// HCI_86620900426153
// MI455X (gfx1250) — matrix operand order repaired and checked by reading, not compiled, not run on hardware
//
#include <hip/hip_runtime.h>
#include <hip/hip_bf16.h>

// ---------------------------------------------------------------------------
// 2-level 2-additive Choquet integral for MI455X (gfx1250, wave32, WMMA).
//
// Roofline: 16 MB of leaf utilities @ 23.3 TB/s => ~0.7us memory floor.
// Strategy:
//   (1) fold max-features away:  CI = sum a'_i x_i + sum (b-c)_ij min(x_i,x_j)
//       (min+max = xi+xj, fold c into the linear term) => 136 features, pad 160.
//   (2) run every dot product on the WMMA pipe (v_wmma_f32_16x16x32_bf16):
//       per 16-row batch tile, chain block-diagonal WMMAs (B fragment of node
//       j nonzero only in column j) so a single 16x16 f32 accumulator ends up
//       holding mid[row][node] in the native C/D layout.
//   (3) occupancy: node columns are disjoint, so 4 waves split the 16-node
//       chain (4 nodes each, 20 chained WMMAs instead of 80) and scatter their
//       disjoint columns into the shared mid tile in LDS => 4096 waves total
//       (~4 per SIMD32 on a ~1024-SIMD part) to hide WMMA/VMEM latency.
//       Root level is one more 5-WMMA replicated-B chain on 1 wave per tile.
// ---------------------------------------------------------------------------

#define NODES   16
#define CH      16
#define NPAIR   120
#define NFEAT   136      // 16 linear + 120 min-pairs
#define KPAD    160      // padded to 5 chunks of K=32 (bf16 WMMA)
#define NCHUNK  5

typedef __attribute__((ext_vector_type(16))) __bf16 v16bf;
typedef __attribute__((ext_vector_type(8)))  float  v8f;

// Upper-triangular pair tables (np.triu_indices(16, k=1) order), constexpr so
// compile-time indices fold to immediates in the fully unrolled hot path.
struct PairTab { int iu[NPAIR]; int ju[NPAIR]; };
constexpr PairTab make_pairs() {
  PairTab t{}; int p = 0;
  for (int i = 0; i < CH; ++i)
    for (int j = i + 1; j < CH; ++j) { t.iu[p] = i; t.ju[p] = j; ++p; }
  return t;
}
constexpr PairTab kPT = make_pairs();

// ---------------------------------------------------------------------------
// Kernel 1: softmax + fold weights.  Output: bf16 wfold[17][KPAD] in d_ws.
//   k <  16 : a'_k = a_k + sum_{pairs containing k} c_p
//   k < 136 : d_{k-16} = b_p - c_p
//   else    : 0 (K padding)
// ---------------------------------------------------------------------------
__global__ void HCI_fold_kernel(const float* __restrict__ w1,
                                const float* __restrict__ wr,
                                __bf16* __restrict__ wfold) {
  __shared__ float sh[256];
  __shared__ float red[256];
  const int node = blockIdx.x;              // 0..15 level-1, 16 = root
  const int t    = threadIdx.x;             // 0..255
  const float* src = (node < NODES) ? (w1 + node * 256) : wr;
  const float v = src[t];

  red[t] = v;
  __syncthreads();
  for (int s = 128; s > 0; s >>= 1) {
    if (t < s) red[t] = fmaxf(red[t], red[t + s]);
    __syncthreads();
  }
  const float mx = red[0];
  __syncthreads();
  const float e = __expf(v - mx);
  red[t] = e;
  __syncthreads();
  for (int s = 128; s > 0; s >>= 1) {
    if (t < s) red[t] += red[t + s];
    __syncthreads();
  }
  const float wn = e / red[0];
  sh[t] = wn;                                // softmax weights: [a(16) b(120) c(120)]
  __syncthreads();

  __bf16* dst = wfold + node * KPAD;
  if (t < CH) {
    float acc = sh[t];
    for (int p = 0; p < NPAIR; ++p)
      if (kPT.iu[p] == t || kPT.ju[p] == t) acc += sh[NFEAT + p];  // + c_p
    dst[t] = (__bf16)acc;
  } else if (t < NFEAT) {
    dst[t] = (__bf16)(sh[t] - sh[t + NPAIR]);                      // b_p - c_p
  } else if (t < KPAD) {
    dst[t] = (__bf16)0.0f;
  }
}

// ---------------------------------------------------------------------------
// Feature value for compile-time K (fully unrolled => xr stays in VGPRs).
// ---------------------------------------------------------------------------
template<int K>
__device__ __forceinline__ float featval(const float xr[CH]) {
  if constexpr (K < CH)         return xr[K];
  else if constexpr (K < NFEAT) return fminf(xr[kPT.iu[K - CH]], xr[kPT.ju[K - CH]]);
  else                          return 0.0f;
}

template<int OFF, int KB>
__device__ __forceinline__ void fill8(v16bf& a, const float xr[CH]) {
  a[OFF + 0] = (__bf16)featval<KB + 0>(xr);
  a[OFF + 1] = (__bf16)featval<KB + 1>(xr);
  a[OFF + 2] = (__bf16)featval<KB + 2>(xr);
  a[OFF + 3] = (__bf16)featval<KB + 3>(xr);
  a[OFF + 4] = (__bf16)featval<KB + 4>(xr);
  a[OFF + 5] = (__bf16)featval<KB + 5>(xr);
  a[OFF + 6] = (__bf16)featval<KB + 6>(xr);
  a[OFF + 7] = (__bf16)featval<KB + 7>(xr);
}

// One K=32 chunk: build bf16 A fragment (contiguous K layout: low lanes
// hold K c*32+0..15, high lanes c*32+16..31, matching B), load the
// B fragment (lanes 0-15 = K 0..15 of chunk, 16-31 = K 16..31; only lanes whose
// column matches `bactive` carry weights), then one chained WMMA accumulate.
template<int C>
__device__ __forceinline__ void ci_chunk(v8f& D, const float xr[CH],
                                         const __bf16* __restrict__ wrow,
                                         bool bactive, bool low) {
  v16bf a;
  if (low) { fill8<0, C * 32 + 0>(a, xr); fill8<8, C * 32 + 8>(a, xr); }
  else     { fill8<0, C * 32 + 16>(a, xr); fill8<8, C * 32 + 24>(a, xr); }

  v16bf b;
#pragma unroll
  for (int t = 0; t < 16; ++t) b[t] = (__bf16)0.0f;
  if (bactive)
    b = *(const v16bf*)(wrow + C * 32 + (low ? 0 : 16));   // 32B, aligned

  // (neg_a, A, neg_b, B, c_mod, C, reuse_a, reuse_b)
  D = __builtin_amdgcn_wmma_f32_16x16x32_bf16(false, a, false, b,
                                              (short)0, D, false, false);
}

// ---------------------------------------------------------------------------
// Kernel 2: main.  4 wave32 per 16-row batch tile (4 nodes each), 8 waves =
// 2 tiles per block.  Disjoint block-diagonal columns let the 4 partial D
// tiles scatter into one LDS mid tile without reduction.
// ---------------------------------------------------------------------------
__global__ __launch_bounds__(256)
void HCI_main_kernel(const float* __restrict__ x,
                     const __bf16* __restrict__ wfold,
                     float* __restrict__ out) {
  __shared__ float lmid[2][16 * 20];        // per-tile 16x16 mid, stride 20 (16B aligned, bank-clean)
  const int tid  = threadIdx.x;
  const int wave = tid >> 5;
  const int lane = tid & 31;
  const bool low = lane < 16;
  const int col  = lane & 15;
  const int lt   = wave >> 2;               // local tile 0..1
  const int quad = wave & 3;                // which 4-node slice this wave owns
  const int tile = blockIdx.x * 2 + lt;
  const int row0 = tile * 16;
  const int row  = row0 + col;              // batch row owned by this lane (A-matrix M)

  __builtin_prefetch(wfold, 0, 1);          // global_prefetch_b8: pin weights near

  // ---- Level 1 slice: nodes quad*4 .. quad*4+3; D columns outside stay 0.
  v8f D = {0.f, 0.f, 0.f, 0.f, 0.f, 0.f, 0.f, 0.f};
  const int j0 = quad * 4;
  for (int j = j0; j < j0 + 4; ++j) {
    const float4* px = (const float4*)(x + (size_t)row * 256 + j * CH);
    const float4 q0 = px[0], q1 = px[1], q2 = px[2], q3 = px[3];
    const float xr[CH] = {q0.x, q0.y, q0.z, q0.w, q1.x, q1.y, q1.z, q1.w,
                          q2.x, q2.y, q2.z, q2.w, q3.x, q3.y, q3.z, q3.w};
    const __bf16* wrow = wfold + j * KPAD;
    const bool act = (col == j);            // B nonzero only in column j
    ci_chunk<0>(D, xr, wrow, act, low);
    ci_chunk<1>(D, xr, wrow, act, low);
    ci_chunk<2>(D, xr, wrow, act, low);
    ci_chunk<3>(D, xr, wrow, act, low);
    ci_chunk<4>(D, xr, wrow, act, low);
  }

  // ---- Scatter this wave's 4 mid columns into the tile's LDS mid matrix.
  // C layout: VGPR r = row r (low lanes) / row r+8 (high lanes), lane = column.
  float* m = lmid[lt];
  if ((col >> 2) == quad) {
#pragma unroll
    for (int r = 0; r < 8; ++r)
      m[(r + (low ? 0 : 8)) * 20 + col] = D[r];
  }
  __syncthreads();

  // ---- Level 2 (root) on 1 wave per tile: replicated-B chain => out[row m]
  if (quad == 0) {
    const float4* pm = (const float4*)(m + col * 20);
    const float4 m0 = pm[0], m1 = pm[1], m2 = pm[2], m3 = pm[3];
    const float xr2[CH] = {m0.x, m0.y, m0.z, m0.w, m1.x, m1.y, m1.z, m1.w,
                           m2.x, m2.y, m2.z, m2.w, m3.x, m3.y, m3.z, m3.w};

    v8f D2 = {0.f, 0.f, 0.f, 0.f, 0.f, 0.f, 0.f, 0.f};
    const __bf16* wroot = wfold + NODES * KPAD;
    ci_chunk<0>(D2, xr2, wroot, true, low);
    ci_chunk<1>(D2, xr2, wroot, true, low);
    ci_chunk<2>(D2, xr2, wroot, true, low);
    ci_chunk<3>(D2, xr2, wroot, true, low);
    ci_chunk<4>(D2, xr2, wroot, true, low);

    // Result replicated across columns; lanes 0 and 16 emit 8 rows each.
    if (col == 0) {
      float4* po = (float4*)(out + row0 + (low ? 0 : 8));
      po[0] = make_float4(D2[0], D2[1], D2[2], D2[3]);
      po[1] = make_float4(D2[4], D2[5], D2[6], D2[7]);
    }
  }
}

// ---------------------------------------------------------------------------
extern "C" void kernel_launch(void* const* d_in, const int* in_sizes, int n_in,
                              void* d_out, int out_size, void* d_ws, size_t ws_size,
                              hipStream_t stream) {
  const float* x  = (const float*)d_in[0];   // [B, 256] f32
  const float* w1 = (const float*)d_in[1];   // [16, 256] f32
  const float* wr = (const float*)d_in[2];   // [256] f32
  float* out      = (float*)d_out;           // [B] f32
  __bf16* wfold   = (__bf16*)d_ws;           // [17][KPAD] bf16 (5440 B)

  const int rows   = in_sizes[0] / 256;      // 16384
  const int ntiles = rows / 16;              // 1024
  const int blocks = ntiles / 2;             // 512 blocks x 256 threads (4 waves/tile)

  HCI_fold_kernel<<<17, 256, 0, stream>>>(w1, wr, wfold);
  HCI_main_kernel<<<blocks, 256, 0, stream>>>(x, wfold, out);
}
